// FragmentsToExpression_1443109012242
// MI455X (gfx1250) — compile-verified, matrix operand order checked by reading, not run on hardware
//
#include <hip/hip_runtime.h>

typedef __attribute__((ext_vector_type(16))) _Float16 v16h;
typedef __attribute__((ext_vector_type(8)))  _Float16 v8h;
typedef __attribute__((ext_vector_type(8)))  float    v8f;

#define WPB   4              // waves per block (wave32)
#define TPB   (WPB * 32)
#define KP    128            // padded K (4 chunks of 32)
#define NP    112            // padded N (7 tiles of 16)
#define NT    7

// v16h WMMA operand overlaid on two 16-byte LDS loads (no element shuffles).
union ABfrag { v16h v; v8h h[2]; };

// One MLP layer: actOut = act(actIn @ W + b), using 16x16x32 f16 WMMA.
// KCNT is compile-time so A-fragments live in statically-indexed registers.
template <int KCNT, bool RELU>
__device__ __attribute__((always_inline)) void mlp_layer(
    const _Float16 (*__restrict__ actIn)[KP],
    _Float16 (*__restrict__ actOut)[KP],
    const _Float16 (*__restrict__ sW)[KP],
    const float* __restrict__ sB,
    int m16, int hi16, int koff)
{
  ABfrag a[KCNT];
  #pragma unroll
  for (int c = 0; c < KCNT; ++c) {
    const _Float16* ap = &actIn[m16][c * 32 + koff];
    a[c].h[0] = *(const v8h*)ap;        // K = c*32+koff .. +7   (ds_load_b128)
    a[c].h[1] = *(const v8h*)(ap + 16); // K = c*32+koff+16 .. +23
  }
  #pragma unroll
  for (int t = 0; t < NT; ++t) {
    const int n = t * 16 + m16;
    v8f acc = {};
    #pragma unroll
    for (int c = 0; c < KCNT; ++c) {
      ABfrag b;
      const _Float16* bp = &sW[n][c * 32 + koff];
      b.h[0] = *(const v8h*)bp;
      b.h[1] = *(const v8h*)(bp + 16);
      acc = __builtin_amdgcn_wmma_f32_16x16x32_f16(
                false, a[c].v, false, b.v, (short)0, acc, false, false);
    }
    // D layout: VGPR r -> row (r + 8*hi16), col n. Bias (+ReLU), back to f16.
    const float bias = sB[n];
    const int mb = hi16 * 8;
    #pragma unroll
    for (int r = 0; r < 8; ++r) {
      float v = acc[r] + bias;
      if (RELU) v = fmaxf(v, 0.0f);
      actOut[mb + r][n] = (_Float16)v;
    }
  }
}

__global__ __launch_bounds__(TPB) void frag_mlp_wmma(
    const int*  __restrict__ coords,
    const int*  __restrict__ cxg_ix,
    const int*  __restrict__ gene_n_p,
    const int*  __restrict__ gene_ix,
    const float* __restrict__ W0, const float* __restrict__ b0,
    const float* __restrict__ W1, const float* __restrict__ b1,
    const float* __restrict__ W2, const float* __restrict__ b2,
    const float* __restrict__ W3, const float* __restrict__ b3,
    const float* __restrict__ weight1,
    float* __restrict__ out,
    int nfrag)
{
  // Transposed, zero-padded weight tile for the current layer: sW[n][k]
  __shared__ __align__(16) _Float16 sW[NP][KP];
  __shared__ float sB[NP];
  // Per-wave ping-pong activation tiles: [16 fragments][128 K] f16
  __shared__ __align__(16) _Float16 sAct[WPB][2][16][KP];

  const int tid  = threadIdx.x;
  const int lane = tid & 31;
  const int wave = tid >> 5;
  const int fragBase = (blockIdx.x * WPB + wave) * 16;

  // Zero activation buffers once (K padding must stay 0 so padded weights
  // never multiply garbage).
  for (int i = tid; i < WPB * 2 * 16 * KP; i += TPB)
    ((_Float16*)sAct)[i] = (_Float16)0.0f;

  // Layer-0 input: x = coords / 1000 into act buffer 0, cols 0..1.
  {
    int m = lane >> 1, col = lane & 1;
    int f = fragBase + m;
    float v = 0.0f;
    if (f < nfrag) v = (float)coords[2 * f + col] * 1.0e-3f;
    sAct[wave][0][m][col] = (_Float16)v;
  }

  const float* Wsrc[4] = {W0, W1, W2, W3};
  const float* Bsrc[4] = {b0, b1, b2, b3};
  const int    Kd[4]   = {2, 100, 100, 100};

  const int m16  = lane & 15;
  const int hi16 = lane >> 4;          // lane group: 0 or 1
  const int koff = hi16 * 8;           // 16-bit A/B layout: +8 K for lanes 16..31

  for (int l = 0; l < 4; ++l) {
    __syncthreads();                   // previous layer's reads of sW complete
    // Stage W_l transposed + zero-padded into LDS (block cooperative).
    const int kd = Kd[l];
    const float* __restrict__ wsrc = Wsrc[l];
    for (int i = tid; i < NP * KP; i += TPB) {
      int n = i >> 7, k = i & (KP - 1);
      float v = (n < 100 && k < kd) ? wsrc[k * 100 + n] : 0.0f;
      sW[n][k] = (_Float16)v;
    }
    for (int n = tid; n < NP; n += TPB)
      sB[n] = (n < 100) ? Bsrc[l][n] : 0.0f;
    __syncthreads();

    // Compile-time K-chunk counts: layer 0 has K=2 -> one 16x16x32 per tile.
    if (l == 0)
      mlp_layer<1, true >(sAct[wave][0], sAct[wave][1], sW, sB, m16, hi16, koff);
    else if (l == 1)
      mlp_layer<4, true >(sAct[wave][1], sAct[wave][0], sW, sB, m16, hi16, koff);
    else if (l == 2)
      mlp_layer<4, true >(sAct[wave][0], sAct[wave][1], sW, sB, m16, hi16, koff);
    else
      mlp_layer<4, false>(sAct[wave][1], sAct[wave][0], sW, sB, m16, hi16, koff);
  }
  __syncthreads();

  // Fused pooling + per-gene projection (emb ended in buffer 0):
  //   out[cxg] += dot(emb[f], weight1[gene_ix[cxg % gene_n]])
  // Lane pair (m16, hi16) splits the E=100 dot; combine via shfl_xor(16).
  {
    const int f = fragBase + m16;
    if (f < nfrag) {
      const int gn  = gene_n_p[0];
      const int cxg = cxg_ix[f];
      const int g   = cxg % gn;
      const float* wrow = weight1 + (long)gene_ix[g] * 100;
      const int e0 = hi16 * 50;
      float acc = 0.0f;
      #pragma unroll 10
      for (int e = 0; e < 50; ++e)
        acc += wrow[e0 + e] * (float)sAct[wave][0][m16][e0 + e];
      acc += __shfl_xor(acc, 16, 32);
      if (hi16 == 0)
        unsafeAtomicAdd(&out[cxg], acc);   // global_atomic_add_f32
    }
  }
}

// Initializes out[c,g] = bias1[gene_ix[g]] every call (d_out is poisoned).
__global__ void bias_init(float* __restrict__ out,
                          const float* __restrict__ bias1,
                          const int* __restrict__ gene_ix,
                          const int* __restrict__ gene_n_p, int total)
{
  int i = blockIdx.x * blockDim.x + threadIdx.x;
  if (i < total) out[i] = bias1[gene_ix[i % gene_n_p[0]]];
}

extern "C" void kernel_launch(void* const* d_in, const int* in_sizes, int n_in,
                              void* d_out, int out_size, void* d_ws, size_t ws_size,
                              hipStream_t stream)
{
  const int*   coords  = (const int*)d_in[0];
  const int*   cxg_ix  = (const int*)d_in[1];
  // d_in[2] = cell_n (unused; out_size = cell_n * gene_n)
  const int*   gene_n  = (const int*)d_in[3];
  const int*   gene_ix = (const int*)d_in[4];
  const float* W0 = (const float*)d_in[5];  const float* b0 = (const float*)d_in[6];
  const float* W1 = (const float*)d_in[7];  const float* b1 = (const float*)d_in[8];
  const float* W2 = (const float*)d_in[9];  const float* b2 = (const float*)d_in[10];
  const float* W3 = (const float*)d_in[11]; const float* b3 = (const float*)d_in[12];
  const float* weight1 = (const float*)d_in[13];
  const float* bias1   = (const float*)d_in[14];
  float* out = (float*)d_out;

  const int nfrag = in_sizes[0] / 2;

  bias_init<<<(out_size + 255) / 256, 256, 0, stream>>>(out, bias1, gene_ix,
                                                        gene_n, out_size);

  const int tiles  = (nfrag + 15) / 16;
  const int blocks = (tiles + WPB - 1) / WPB;
  frag_mlp_wmma<<<blocks, TPB, 0, stream>>>(coords, cxg_ix, gene_n, gene_ix,
                                            W0, b0, W1, b1, W2, b2, W3, b3,
                                            weight1, out, nfrag);
}
